// Head_7799660610344
// MI455X (gfx1250) — matrix operand order repaired and checked by reading, not compiled, not run on hardware
//
#include <hip/hip_runtime.h>

// ---- sizes (fixed by the reference) ----
#define BATCH 8
#define TSEQ  2048
#define CEMB  1024
#define HD    128
#define MROWS (BATCH * TSEQ)   // 16384
#define SCALE 0.03125f         // 1/sqrt(1024) -- reference scales by n_embd

// ---- WMMA fragment types (CDNA5 wave32) ----
typedef __bf16 bf16x8  __attribute__((ext_vector_type(8)));
typedef __bf16 bf16x16 __attribute__((ext_vector_type(16)));
typedef float  f32x8   __attribute__((ext_vector_type(8)));

static __device__ inline bf16x16 ld_bf16x16(const __bf16* p) {
    bf16x8 lo = *(const bf16x8*)p;
    bf16x8 hi = *(const bf16x8*)(p + 16);
    bf16x16 r;
#pragma unroll
    for (int i = 0; i < 8; ++i) { r[i] = lo[i]; r[8 + i] = hi[i]; }
    return r;
}

static __device__ inline f32x8 f32x8_zero() {
    f32x8 z = {0.f, 0.f, 0.f, 0.f, 0.f, 0.f, 0.f, 0.f};
    return z;
}

// CDNA5 async copy: per-lane LDS[ldsaddr] = MEM[gaddr] (16B), tracked by ASYNCcnt.
static __device__ inline void async_copy_b128(unsigned ldsaddr, const void* gaddr) {
    asm volatile("global_load_async_to_lds_b128 %0, %1, off"
                 :: "v"(ldsaddr), "v"((unsigned long long)(uintptr_t)gaddr)
                 : "memory");
}
static __device__ inline void wait_asynccnt0() {
    asm volatile("s_wait_asynccnt 0x0" ::: "memory");
}

// ============================================================
// Kernel 0: pack W [C,H] f32 -> Wt [H,C] bf16 (3 matrices)
// ============================================================
__global__ __launch_bounds__(256) void pack_wt_kernel(
    const float* __restrict__ Wq, const float* __restrict__ Wk,
    const float* __restrict__ Wv, __bf16* __restrict__ wt)
{
    const int idx = blockIdx.x * 256 + threadIdx.x;       // < 3*128*1024
    const int mat = idx >> 17;
    const int rem = idx & 131071;
    const int h = rem >> 10;
    const int c = rem & 1023;
    const float* W = (mat == 0) ? Wq : ((mat == 1) ? Wk : Wv);
    wt[(size_t)mat * (HD * CEMB) + (size_t)h * CEMB + c] = (__bf16)W[(size_t)c * HD + h];
}

// ============================================================
// Kernel 1: projection  out[M,128] = bf16(x[M,1024] @ W + b)
// 256 threads = 8 waves; block computes 64 rows x 128 cols.
// True ping-pong pipeline: two fixed buffer slots, two chunks per
// iteration, zero register rotation moves.
// ============================================================
__global__ __launch_bounds__(256) void qkv_proj_kernel(
    const float* __restrict__ x, const __bf16* __restrict__ wt,
    const float* __restrict__ bias, __bf16* __restrict__ out)
{
    const int tid  = threadIdx.x;
    const int wave = tid >> 5;
    const int lane = tid & 31;
    const int hf   = lane >> 4;          // half-wave select
    const int lm   = lane & 15;
    const int kh   = hf * 8;             // A-frag K sub-offset

    const int rowBase = blockIdx.x * 64 + (wave >> 1) * 16;
    const int colBase = (wave & 1) * 64;

    f32x8 acc[4];
#pragma unroll
    for (int t = 0; t < 4; ++t) acc[t] = f32x8_zero();

    const float*  xrow = x + (size_t)(rowBase + lm) * CEMB;
    const __bf16* wrow[4];
#pragma unroll
    for (int t = 0; t < 4; ++t)
        wrow[t] = wt + (size_t)(colBase + t * 16 + lm) * CEMB + hf * 8;

    // slot 0 holds chunk c, slot 1 holds chunk c+32
    f32x8   aL[2], aH[2];
    bf16x16 bB[2][4];

    // prologue: chunk 0 -> slot 0
    aL[0] = *(const f32x8*)(xrow + kh);
    aH[0] = *(const f32x8*)(xrow + 16 + kh);
#pragma unroll
    for (int t = 0; t < 4; ++t) bB[0][t] = ld_bf16x16(wrow[t]);

    for (int c = 0; c < CEMB; c += 64) {
        // issue chunk c+32 -> slot 1 (always in range: c+32 <= 992)
        aL[1] = *(const f32x8*)(xrow + c + 32 + kh);
        aH[1] = *(const f32x8*)(xrow + c + 32 + 16 + kh);
#pragma unroll
        for (int t = 0; t < 4; ++t) bB[1][t] = ld_bf16x16(wrow[t] + c + 32);

        // compute slot 0 (chunk c)
        bf16x16 a0;
#pragma unroll
        for (int i = 0; i < 8; ++i) { a0[i] = (__bf16)aL[0][i]; a0[8 + i] = (__bf16)aH[0][i]; }
#pragma unroll
        for (int t = 0; t < 4; ++t)
            acc[t] = __builtin_amdgcn_wmma_f32_16x16x32_bf16(
                false, a0, false, bB[0][t], (short)0, acc[t], false, false);

        // issue chunk c+64 -> slot 0
        if (c + 64 < CEMB) {
            aL[0] = *(const f32x8*)(xrow + c + 64 + kh);
            aH[0] = *(const f32x8*)(xrow + c + 64 + 16 + kh);
#pragma unroll
            for (int t = 0; t < 4; ++t) bB[0][t] = ld_bf16x16(wrow[t] + c + 64);
        }

        // compute slot 1 (chunk c+32)
        bf16x16 a1;
#pragma unroll
        for (int i = 0; i < 8; ++i) { a1[i] = (__bf16)aL[1][i]; a1[8 + i] = (__bf16)aH[1][i]; }
#pragma unroll
        for (int t = 0; t < 4; ++t)
            acc[t] = __builtin_amdgcn_wmma_f32_16x16x32_bf16(
                false, a1, false, bB[1][t], (short)0, acc[t], false, false);
    }

    // bias + store bf16 (C layout: row = v + 8*hf, col = lm)
#pragma unroll
    for (int t = 0; t < 4; ++t) {
        const int col = colBase + t * 16 + lm;
        const float bv = bias[col];
#pragma unroll
        for (int v = 0; v < 8; ++v) {
            const int row = rowBase + v + 8 * hf;
            out[(size_t)row * HD + col] = (__bf16)(acc[t][v] + bv);
        }
    }
}

// ============================================================
// Kernel 2: flash attention (causal, online softmax)
// 128 threads = 4 waves; block = (batch, 64 query rows); wave = 16 rows.
// Double-buffered KV tiles: async-to-LDS K(j+1) + VGPR-held V(j+1)
// overlap with compute on tile j. One asynccnt wait + barrier per step.
// ============================================================
#define K_LD  136   // 272B rows: 16B aligned, 4-bank lane stride
#define VT_LD 72    // 144B rows: 16B aligned, 36-bank lane stride
#define P_LD  72

__global__ __launch_bounds__(128) void flash_attn_kernel(
    const __bf16* __restrict__ qg, const __bf16* __restrict__ kg,
    const __bf16* __restrict__ vg, float* __restrict__ out)
{
    __shared__ __bf16 Ks[2][64 * K_LD];
    __shared__ __bf16 Vt[2][HD * VT_LD];
    __shared__ __bf16 Pst[4 * 16 * P_LD];

    const int tid   = threadIdx.x;
    const int wave  = tid >> 5;
    const int lane  = tid & 31;
    const int hf    = lane >> 4;
    const int lm    = lane & 15;
    const int kh    = hf * 8;
    const int qb    = blockIdx.x;            // query block [0,32)
    const int batch = blockIdx.y;

    // --- load this wave's Q fragments (16 rows x 128 h) straight from global
    bf16x16 qf[4];
    const __bf16* qrow = qg + ((size_t)(batch * TSEQ + qb * 64 + wave * 16 + lm)) * HD;
#pragma unroll
    for (int f = 0; f < 4; ++f) {
        bf16x8 lo = *(const bf16x8*)(qrow + f * 32 + kh);
        bf16x8 hi = *(const bf16x8*)(qrow + f * 32 + 16 + kh);
#pragma unroll
        for (int i = 0; i < 8; ++i) { qf[f][i] = lo[i]; qf[f][8 + i] = hi[i]; }
    }

    f32x8 o[8];
#pragma unroll
    for (int t = 0; t < 8; ++t) o[t] = f32x8_zero();
    float m_run[8], l_run[8];
#pragma unroll
    for (int v = 0; v < 8; ++v) { m_run[v] = -1e30f; l_run[v] = 0.f; }

    const int hc    = tid & 15;              // h chunk (8 elems) for coop loads
    const int krow0 = tid >> 4;              // 0..7

    // ---- prologue: stage tile 0 into buffer 0
    {
        const size_t tbase = ((size_t)(batch * TSEQ)) * HD;
#pragma unroll
        for (int pass = 0; pass < 8; ++pass) {
            const int key = krow0 + pass * 8;
            const size_t src = tbase + (size_t)key * HD + hc * 8;
            async_copy_b128((unsigned)(uintptr_t)&Ks[0][key * K_LD + hc * 8], kg + src);
            bf16x8 vv = *(const bf16x8*)(vg + src);
#pragma unroll
            for (int i = 0; i < 8; ++i) Vt[0][(hc * 8 + i) * VT_LD + key] = vv[i];
        }
        wait_asynccnt0();
        __syncthreads();
    }

    for (int j = 0; j <= qb; ++j) {
        const int cb = j & 1;
        const int nb = cb ^ 1;
        const bool more = (j < qb);

        // ---- issue next tile: async K(j+1) -> Ks[nb], V(j+1) -> VGPRs
        bf16x8 vnx[8];
        if (more) {
            const size_t tnext = ((size_t)(batch * TSEQ + (j + 1) * 64)) * HD;
#pragma unroll
            for (int pass = 0; pass < 8; ++pass) {
                const int key = krow0 + pass * 8;
                const size_t src = tnext + (size_t)key * HD + hc * 8;
                async_copy_b128((unsigned)(uintptr_t)&Ks[nb][key * K_LD + hc * 8], kg + src);
                vnx[pass] = *(const bf16x8*)(vg + src);
            }
        }
        if (j + 2 <= qb) { // warm L2 two tiles ahead (global_prefetch_b8)
            const size_t t2 = ((size_t)(batch * TSEQ + (j + 2) * 64)) * HD + (size_t)tid * 64;
            __builtin_prefetch(kg + t2, 0, 1);
            __builtin_prefetch(vg + t2, 0, 1);
        }

        // ---- S = Q @ K^T on current buffer (4 key tiles, contract h=128)
        f32x8 s[4];
#pragma unroll
        for (int t = 0; t < 4; ++t) s[t] = f32x8_zero();
#pragma unroll
        for (int f = 0; f < 4; ++f) {
#pragma unroll
            for (int t = 0; t < 4; ++t) {
                bf16x16 b = ld_bf16x16(&Ks[cb][(t * 16 + lm) * K_LD + f * 32 + hf * 8]);
                s[t] = __builtin_amdgcn_wmma_f32_16x16x32_bf16(
                    false, qf[f], false, b, (short)0, s[t], false, false);
            }
        }

        // ---- scatter V(j+1) into Vt[nb] (overlaps with softmax below)
        if (more) {
#pragma unroll
            for (int pass = 0; pass < 8; ++pass) {
                const int key = krow0 + pass * 8;
#pragma unroll
                for (int i = 0; i < 8; ++i) Vt[nb][(hc * 8 + i) * VT_LD + key] = vnx[pass][i];
            }
        }

        // ---- scale + causal mask (diagonal block only)
        if (j == qb) {
#pragma unroll
            for (int t = 0; t < 4; ++t)
#pragma unroll
                for (int v = 0; v < 8; ++v) {
                    const int irow = wave * 16 + v + 8 * hf;
                    const int jcol = t * 16 + lm;
                    s[t][v] = (jcol > irow) ? -1e30f : s[t][v] * SCALE;
                }
        } else {
#pragma unroll
            for (int t = 0; t < 4; ++t)
#pragma unroll
                for (int v = 0; v < 8; ++v) s[t][v] *= SCALE;
        }

        // ---- online softmax: each row lives in one VGPR across a 16-lane half
#pragma unroll
        for (int v = 0; v < 8; ++v) {
            float rm = fmaxf(fmaxf(s[0][v], s[1][v]), fmaxf(s[2][v], s[3][v]));
            rm = fmaxf(rm, __shfl_xor(rm, 1, 32));
            rm = fmaxf(rm, __shfl_xor(rm, 2, 32));
            rm = fmaxf(rm, __shfl_xor(rm, 4, 32));
            rm = fmaxf(rm, __shfl_xor(rm, 8, 32));
            const float mnew  = fmaxf(m_run[v], rm);
            const float alpha = __expf(m_run[v] - mnew);
            m_run[v] = mnew;
            float rs = 0.f;
#pragma unroll
            for (int t = 0; t < 4; ++t) { s[t][v] = __expf(s[t][v] - mnew); rs += s[t][v]; }
            rs += __shfl_xor(rs, 1, 32);
            rs += __shfl_xor(rs, 2, 32);
            rs += __shfl_xor(rs, 4, 32);
            rs += __shfl_xor(rs, 8, 32);
            l_run[v] = l_run[v] * alpha + rs;
#pragma unroll
            for (int t = 0; t < 8; ++t) o[t][v] *= alpha;
        }

        // ---- P (C layout) -> LDS -> A layout (same-wave DS ops are in-order)
        __bf16* pw = &Pst[wave * 16 * P_LD];
#pragma unroll
        for (int t = 0; t < 4; ++t)
#pragma unroll
            for (int v = 0; v < 8; ++v)
                pw[(v + 8 * hf) * P_LD + t * 16 + lm] = (__bf16)s[t][v];

        // ---- O += P @ V on current buffer (contract 64 keys = 2 K-steps)
#pragma unroll
        for (int f2 = 0; f2 < 2; ++f2) {
            const __bf16* pr = pw + lm * P_LD + f2 * 32 + kh;
            bf16x8 lo = *(const bf16x8*)pr;
            bf16x8 hi = *(const bf16x8*)(pr + 16);
            bf16x16 ap;
#pragma unroll
            for (int i = 0; i < 8; ++i) { ap[i] = lo[i]; ap[8 + i] = hi[i]; }
#pragma unroll
            for (int ot = 0; ot < 8; ++ot) {
                bf16x16 bv = ld_bf16x16(&Vt[cb][(ot * 16 + lm) * VT_LD + f2 * 32 + hf * 8]);
                o[ot] = __builtin_amdgcn_wmma_f32_16x16x32_bf16(
                    false, ap, false, bv, (short)0, o[ot], false, false);
            }
        }

        wait_asynccnt0();       // K(j+1) resident in Ks[nb]
        __syncthreads();        // all waves done reading cb, done writing nb
    }

    // ---- epilogue: out = O / l  (f32)
    float* op = out + ((size_t)(batch * TSEQ + qb * 64 + wave * 16)) * HD;
#pragma unroll
    for (int ot = 0; ot < 8; ++ot) {
        const int col = ot * 16 + lm;
#pragma unroll
        for (int v = 0; v < 8; ++v) {
            const int row = v + 8 * hf;
            op[(size_t)row * HD + col] = o[ot][v] / l_run[v];
        }
    }
}

// ============================================================
// launch
// ============================================================
extern "C" void kernel_launch(void* const* d_in, const int* in_sizes, int n_in,
                              void* d_out, int out_size, void* d_ws, size_t ws_size,
                              hipStream_t stream) {
    const float* x  = (const float*)d_in[0];
    const float* Wq = (const float*)d_in[1];
    const float* bq = (const float*)d_in[2];
    const float* Wk = (const float*)d_in[3];
    const float* bk = (const float*)d_in[4];
    const float* Wv = (const float*)d_in[5];
    const float* bv = (const float*)d_in[6];
    float* out = (float*)d_out;

    // workspace layout: 3x Wt (bf16, 128x1024) then 3x QKV (bf16, 16384x128)
    __bf16* wt  = (__bf16*)d_ws;
    __bf16* wtq = wt;
    __bf16* wtk = wt + (size_t)HD * CEMB;
    __bf16* wtv = wt + (size_t)2 * HD * CEMB;
    __bf16* qw  = wt + (size_t)3 * HD * CEMB;
    __bf16* kw  = qw + (size_t)MROWS * HD;
    __bf16* vw  = kw + (size_t)MROWS * HD;

    pack_wt_kernel<<<dim3((3 * HD * CEMB) / 256), 256, 0, stream>>>(Wq, Wk, Wv, wt);

    qkv_proj_kernel<<<dim3(MROWS / 64), 256, 0, stream>>>(x, wtq, bq, qw);
    qkv_proj_kernel<<<dim3(MROWS / 64), 256, 0, stream>>>(x, wtk, bk, kw);
    qkv_proj_kernel<<<dim3(MROWS / 64), 256, 0, stream>>>(x, wtv, bv, vw);

    flash_attn_kernel<<<dim3(TSEQ / 64, BATCH), 128, 0, stream>>>(qw, kw, vw, out);
}
